// ScaledDotProductAttention_42640435315043
// MI455X (gfx1250) — matrix operand order repaired and checked by reading, not compiled, not run on hardware
//
#include <hip/hip_runtime.h>

// CDNA5 / gfx1250 masked scaled-dot-product attention.
//   q,k,v : (8, 2048, 256) fp32     adj : (8, 2048, 2048) int32
//   out   : output (8*2048, 256) fp32  ++  attn (8, 2048, 2048) fp32
//
// One workgroup (256 threads = 8 wave32) handles one (batch, 16-row) tile.
// Scores for the whole 16 x 2048 strip live in LDS (never round-trip HBM);
// attn is written exactly once, adj read exactly once -> ~335 MB total HBM
// traffic, the bandwidth floor for this problem.

typedef __attribute__((ext_vector_type(16))) _Float16 v16h;
typedef __attribute__((ext_vector_type(8)))  float    v8f;

#define B_SZ    8
#define N_SZ    2048
#define D_SZ    256
#define TILE    16
#define NTHR    256
#define NWAVE   8
#define COLT    (N_SZ / TILE)          // 128 column tiles of 16
#define KCH     (D_SZ / 32)            // 8 K-chunks of 32 for QK^T
#define PCH     (N_SZ / 32)            // 64 K-chunks of 32 for P*V
#define TEMP_INV   (1.0f / 16.0f)
#define MASK_FILLV (-1.0e9f)

// Dynamic-LDS layout (bytes). Score rows padded 2048->2052 floats so the
// per-lane row stride is 4 banks (not 0) -> conflict-free fragment loads.
#define SROW       2052                              // floats per score row
#define QROW       264                               // halves per Q row (pad)
#define SC_BYTES   (TILE * SROW * 4)                 // 131328
#define Q_OFF      SC_BYTES                          // 16B aligned
#define Q_BYTES    (TILE * QROW * 2)                 // 8448
#define V_OFF      (Q_OFF + Q_BYTES)                 // 139776, 16B aligned
#define V_BYTES    (32 * D_SZ * 2)                   // 16384
#define LDS_BYTES  (V_OFF + V_BYTES)                 // 156160

__launch_bounds__(NTHR)
__global__ void attn_wmma_kernel(const float* __restrict__ q,
                                 const float* __restrict__ k,
                                 const float* __restrict__ v,
                                 const int*   __restrict__ adj,
                                 float* __restrict__ out,
                                 float* __restrict__ attn)
{
    extern __shared__ char smem_raw[];
    float*    scores = reinterpret_cast<float*>(smem_raw);
    _Float16* qtile  = reinterpret_cast<_Float16*>(smem_raw + Q_OFF);
    _Float16* vtile  = reinterpret_cast<_Float16*>(smem_raw + V_OFF);
    __shared__ float red[NTHR];
    __shared__ float rowinv[TILE];

    const int tid  = threadIdx.x;
    const int lane = tid & 31;
    const int wave = tid >> 5;
    const int half = lane >> 4;    // 0: lanes 0-15, 1: lanes 16-31
    const int m    = lane & 15;

    const int b  = blockIdx.x / COLT;
    const int it = blockIdx.x % COLT;
    const int i0 = it * TILE;

    const size_t qkv_base = (size_t)b * N_SZ * D_SZ;
    const size_t adj_base = (size_t)b * N_SZ * N_SZ;

    // ---- Phase 0: stage 16x256 Q tile to LDS as f16 (coalesced) ----
    {
        const int row  = tid >> 4;
        const int col0 = (tid & 15) * 16;
        const float* src = q + qkv_base + (size_t)(i0 + row) * D_SZ + col0;
        _Float16*    dst = qtile + row * QROW + col0;
#pragma unroll
        for (int i = 0; i < 16; ++i) dst[i] = (_Float16)src[i];
    }
    __syncthreads();

    // ---- Per-wave Q A-fragments (16-bit A 16x32, contiguous k order):
    // lane<16 holds K = [c..c+15]; lane>=16 holds [c+16..c+31]. ----
    v16h aq[KCH];
#pragma unroll
    for (int kc = 0; kc < KCH; ++kc) {
        const _Float16* qr = qtile + m * QROW + kc * 32 + half * 16;
        v16h a;
#pragma unroll
        for (int e = 0; e < 8; ++e) { a[e] = qr[e]; a[8 + e] = qr[8 + e]; }
        aq[kc] = a;
    }

    // ---- Phase 1: S = (Q/T) K^T with mask, strip parked in LDS ----
    for (int jt = wave; jt < COLT; jt += NWAVE) {
        const int j0 = jt * TILE;
        const float* krow = k + qkv_base + (size_t)(j0 + m) * D_SZ;
        if (jt + NWAVE < COLT)  // global_prefetch_b8 for the next K tile
            __builtin_prefetch(k + qkv_base + (size_t)(j0 + NWAVE * TILE + m) * D_SZ, 0, 1);

        v8f acc = {};
#pragma unroll
        for (int kc = 0; kc < KCH; ++kc) {
            // B 32x16: lane holds column N=m; lanes 0-15 rows K 0-15,
            // lanes 16-31 rows K 16-31 of this chunk (column-major source).
            const float* src = krow + kc * 32 + half * 16;
            v16h bf;
#pragma unroll
            for (int e = 0; e < 16; ++e) bf[e] = (_Float16)src[e];
            acc = __builtin_amdgcn_wmma_f32_16x16x32_f16(
                      false, aq[kc], false, bf, (short)0, acc, false, false);
        }

        // C/D layout: VGPR r, lane L -> M = r + (L/16)*8, N = L%16
        const int gj = j0 + m;
#pragma unroll
        for (int r = 0; r < 8; ++r) {
            const int row = r + half * 8;
            const int av  = adj[adj_base + (size_t)(i0 + row) * N_SZ + gj];
            const float s = (av == 0) ? MASK_FILLV : acc[r] * TEMP_INV;
            scores[row * SROW + j0 + m] = s;
        }
    }
    __syncthreads();

    // ---- Phase 2: softmax over each of the 16 rows (all in LDS) ----
    {
        const int prow = tid >> 4;        // 0..15
        const int pseg = tid & 15;        // interleaved columns: pseg + 16*i
        float* srow = scores + prow * SROW;

        float mloc = -3.0e38f;
        for (int i = 0; i < 128; ++i)
            mloc = fmaxf(mloc, srow[pseg + 16 * i]);
        red[tid] = mloc;
        __syncthreads();
        float rmax = -3.0e38f;
#pragma unroll
        for (int i = 0; i < 16; ++i) rmax = fmaxf(rmax, red[prow * 16 + i]);
        __syncthreads();

        float sloc = 0.0f;
        for (int i = 0; i < 128; ++i) {
            const int c = pseg + 16 * i;
            const float e = __expf(srow[c] - rmax);
            srow[c] = e;
            sloc += e;
        }
        red[tid] = sloc;
        __syncthreads();
        if (tid < TILE) {
            float rs = 0.0f;
#pragma unroll
            for (int i = 0; i < 16; ++i) rs += red[tid * 16 + i];
            rowinv[tid] = 1.0f / rs;
        }
        __syncthreads();

        // Normalize in LDS and stream attn to HBM, fully coalesced float4.
        for (int r = 0; r < TILE; ++r) {
            const float inv = rowinv[r];
            float* sr = scores + r * SROW;
            float* ar = attn + ((size_t)b * N_SZ + i0 + r) * N_SZ;
            for (int c = tid * 4; c < N_SZ; c += NTHR * 4) {
                float4 pv;
                pv.x = sr[c + 0] * inv; pv.y = sr[c + 1] * inv;
                pv.z = sr[c + 2] * inv; pv.w = sr[c + 3] * inv;
                *reinterpret_cast<float4*>(sr + c) = pv;   // keep P for phase 3
                *reinterpret_cast<float4*>(ar + c) = pv;   // 134 MB stream, once
            }
        }
    }
    __syncthreads();

    // ---- Phase 3: O = P * V, V staged chunk-wise through LDS ----
    v8f oacc0 = {}, oacc1 = {};
    const int vt0 = wave * 2, vt1 = wave * 2 + 1;   // two 16-col output tiles
    const int vcol0 = vt0 * TILE + m;
    const int vcol1 = vt1 * TILE + m;

    for (int kc2 = 0; kc2 < PCH; ++kc2) {
        {   // cooperative stage of V rows [kc2*32, kc2*32+32) x 256, coalesced
            const int rr = tid >> 3;
            const int c0 = (tid & 7) * 32;
            const float* src = v + qkv_base + (size_t)(kc2 * 32 + rr) * D_SZ + c0;
            _Float16*    dst = vtile + rr * D_SZ + c0;
#pragma unroll
            for (int i = 0; i < 32; ++i) dst[i] = (_Float16)src[i];
        }
        __syncthreads();

        // A-fragment from P (fp32 in LDS -> f16), same 16x32 layout as aq
        const float* pr = scores + m * SROW + kc2 * 32 + half * 16;
        v16h ap;
#pragma unroll
        for (int e = 0; e < 8; ++e) {
            ap[e]     = (_Float16)pr[e];
            ap[8 + e] = (_Float16)pr[8 + e];
        }
        // B-fragments from staged V chunk (lane = one V column)
        const _Float16* vb = vtile + (half * 16) * D_SZ;
        v16h bv0, bv1;
#pragma unroll
        for (int e = 0; e < 16; ++e) {
            bv0[e] = vb[e * D_SZ + vcol0];
            bv1[e] = vb[e * D_SZ + vcol1];
        }
        oacc0 = __builtin_amdgcn_wmma_f32_16x16x32_f16(
                    false, ap, false, bv0, (short)0, oacc0, false, false);
        oacc1 = __builtin_amdgcn_wmma_f32_16x16x32_f16(
                    false, ap, false, bv1, (short)0, oacc1, false, false);
        __syncthreads();
    }

    // ---- Write the 16x256 output tile (64B per half-wave row, coalesced) ----
#pragma unroll
    for (int r = 0; r < 8; ++r) {
        const int row = r + half * 8;
        float* orow = out + ((size_t)b * N_SZ + i0 + row) * D_SZ;
        orow[vt0 * TILE + m] = oacc0[r];
        orow[vt1 * TILE + m] = oacc1[r];
    }
}

extern "C" void kernel_launch(void* const* d_in, const int* in_sizes, int n_in,
                              void* d_out, int out_size, void* d_ws, size_t ws_size,
                              hipStream_t stream) {
    (void)in_sizes; (void)n_in; (void)out_size; (void)d_ws; (void)ws_size;
    const float* q   = (const float*)d_in[0];
    const float* k   = (const float*)d_in[1];
    const float* v   = (const float*)d_in[2];
    const int*   adj = (const int*)d_in[3];
    float* out  = (float*)d_out;                          // (B*n, d)
    float* attn = out + (size_t)B_SZ * N_SZ * D_SZ;       // (B, n, n)

    // >64KB dynamic LDS opt-in (idempotent; capture-safe, not a stream op).
    hipFuncSetAttribute(reinterpret_cast<const void*>(attn_wmma_kernel),
                        hipFuncAttributeMaxDynamicSharedMemorySize, LDS_BYTES);

    dim3 grid(B_SZ * COLT);   // 8 * 128 = 1024 workgroups
    dim3 block(NTHR);         // 8 wave32
    attn_wmma_kernel<<<grid, block, LDS_BYTES, stream>>>(q, k, v, adj, out, attn);
}
